// HyenaRegressor_MLP_cat_ReducedRelu_42150809043565
// MI455X (gfx1250) — matrix operand order repaired and checked by reading, not compiled, not run on hardware
//
#include <hip/hip_runtime.h>

// ---------------------------------------------------------------------------
// HyenaRegressor forward for MI455X (gfx1250, wave32, WMMA).
//   GEMMs   : v_wmma_f32_16x16x32_bf16 (bf16 in, f32 accumulate)
//             - fast kernel: fully tile-aligned, no guards (layer GEMMs)
//             - edge kernel: branch-free clamped loads (head GEMMs)
//   FFT conv: radix-2 in-LDS FFT, N=16384 complex, 128KB LDS/WG
//             (CDNA5 320KB/WGP), fed by global_load_async_to_lds_b128
// ---------------------------------------------------------------------------

#define BB   16
#define LL   8192
#define DD   192
#define HH   3
#define HD   64
#define CIN  576           // (ORDER+1)*D
#define NSEQ 16384         // 2*L FFT length
#define ML   (BB*LL)       // 131072 rows

typedef __attribute__((ext_vector_type(16))) __bf16 v16bf;
typedef __attribute__((ext_vector_type(8)))  float  v8f;

#if defined(__has_builtin)
#if __has_builtin(__builtin_amdgcn_s_wait_asynccnt)
#define WAIT_ASYNC0() __builtin_amdgcn_s_wait_asynccnt(0)
#else
#define WAIT_ASYNC0() asm volatile("s_wait_asynccnt 0x0" ::: "memory")
#endif
#else
#define WAIT_ASYNC0() asm volatile("s_wait_asynccnt 0x0" ::: "memory")
#endif

// Async DMA: 16 bytes global -> LDS (per lane). ldsoff = low 32 bits of the
// generic shared-pointer address.
__device__ __forceinline__ void async_g2l_b128(unsigned ldsoff, const float* g) {
  asm volatile("global_load_async_to_lds_b128 %0, %1, off"
               :: "v"(ldsoff), "v"(g) : "memory");
}

// ---------------------------------------------------------------------------
// Fragment loaders. ISA layouts:
// A 16x32: lane<16 -> M=lane, K = {kb..kb+7, kb+16..kb+23}, kb = k0+8*(lane>=16)
// B 32x16: lane%16 = N, K = {kb..kb+7, kb+16..kb+23}, kb = k0+8*(lane>=16) (same as A)
// ---------------------------------------------------------------------------
__device__ __forceinline__ v16bf cvt16(float4 f0, float4 f1, float4 f2, float4 f3) {
  v16bf f;
  f[0]=(__bf16)f0.x; f[1]=(__bf16)f0.y; f[2]=(__bf16)f0.z; f[3]=(__bf16)f0.w;
  f[4]=(__bf16)f1.x; f[5]=(__bf16)f1.y; f[6]=(__bf16)f1.z; f[7]=(__bf16)f1.w;
  f[8]=(__bf16)f2.x; f[9]=(__bf16)f2.y; f[10]=(__bf16)f2.z; f[11]=(__bf16)f2.w;
  f[12]=(__bf16)f3.x; f[13]=(__bf16)f3.y; f[14]=(__bf16)f3.z; f[15]=(__bf16)f3.w;
  return f;
}

// ---- fast path: no bounds checks (caller guarantees alignment) ----
__device__ __forceinline__ v16bf load_fragA_fast(const float* __restrict__ X,
                                                 int K, int r, int kb) {
  const float* p = X + (size_t)r * K + kb;
  return cvt16(*(const float4*)(p),      *(const float4*)(p + 4),
               *(const float4*)(p + 16), *(const float4*)(p + 20));
}
__device__ __forceinline__ v16bf load_fragB_fast(const float* __restrict__ W,
                                                 int K, int n, int kb) {
  const float* p = W + (size_t)n * K + kb;
  return cvt16(*(const float4*)(p),     *(const float4*)(p + 4),
               *(const float4*)(p + 16), *(const float4*)(p + 20));
}

// ---- edge path: branch-free clamp + select (requires K%8==0, block starts %8)
__device__ __forceinline__ v16bf load_fragA_edge(const float* __restrict__ X,
                                                 int rows, int K, int r, int kb) {
  bool rok = r < rows;
  int rc = rok ? r : 0;
  const float* p = X + (size_t)rc * K;
  int s0 = kb;      if (s0 > K - 8) s0 = K - 8;
  int s1 = kb + 16; if (s1 > K - 8) s1 = K - 8;
  float4 f0 = *(const float4*)(p + s0);
  float4 f1 = *(const float4*)(p + s0 + 4);
  float4 f2 = *(const float4*)(p + s1);
  float4 f3 = *(const float4*)(p + s1 + 4);
  bool ok0 = rok && (kb < K);
  bool ok1 = rok && (kb + 16 < K);
  v16bf f;
  f[0]=(__bf16)(ok0?f0.x:0.f); f[1]=(__bf16)(ok0?f0.y:0.f);
  f[2]=(__bf16)(ok0?f0.z:0.f); f[3]=(__bf16)(ok0?f0.w:0.f);
  f[4]=(__bf16)(ok0?f1.x:0.f); f[5]=(__bf16)(ok0?f1.y:0.f);
  f[6]=(__bf16)(ok0?f1.z:0.f); f[7]=(__bf16)(ok0?f1.w:0.f);
  f[8]=(__bf16)(ok1?f2.x:0.f); f[9]=(__bf16)(ok1?f2.y:0.f);
  f[10]=(__bf16)(ok1?f2.z:0.f); f[11]=(__bf16)(ok1?f2.w:0.f);
  f[12]=(__bf16)(ok1?f3.x:0.f); f[13]=(__bf16)(ok1?f3.y:0.f);
  f[14]=(__bf16)(ok1?f3.z:0.f); f[15]=(__bf16)(ok1?f3.w:0.f);
  return f;
}
__device__ __forceinline__ v16bf load_fragB_edge(const float* __restrict__ W,
                                                 int rows, int K, int n, int kb) {
  bool nok = n < rows;
  int nc = nok ? n : 0;
  const float* p = W + (size_t)nc * K;
  int s0 = kb;     if (s0 > K - 8) s0 = K - 8;
  int s1 = kb + 16; if (s1 > K - 8) s1 = K - 8;
  float4 f0 = *(const float4*)(p + s0);
  float4 f1 = *(const float4*)(p + s0 + 4);
  float4 f2 = *(const float4*)(p + s1);
  float4 f3 = *(const float4*)(p + s1 + 4);
  bool ok0 = nok && (kb < K);
  bool ok1 = nok && (kb + 16 < K);
  v16bf f;
  f[0]=(__bf16)(ok0?f0.x:0.f); f[1]=(__bf16)(ok0?f0.y:0.f);
  f[2]=(__bf16)(ok0?f0.z:0.f); f[3]=(__bf16)(ok0?f0.w:0.f);
  f[4]=(__bf16)(ok0?f1.x:0.f); f[5]=(__bf16)(ok0?f1.y:0.f);
  f[6]=(__bf16)(ok0?f1.z:0.f); f[7]=(__bf16)(ok0?f1.w:0.f);
  f[8]=(__bf16)(ok1?f2.x:0.f); f[9]=(__bf16)(ok1?f2.y:0.f);
  f[10]=(__bf16)(ok1?f2.z:0.f); f[11]=(__bf16)(ok1?f2.w:0.f);
  f[12]=(__bf16)(ok1?f3.x:0.f); f[13]=(__bf16)(ok1?f3.y:0.f);
  f[14]=(__bf16)(ok1?f3.z:0.f); f[15]=(__bf16)(ok1?f3.w:0.f);
  return f;
}

// ---------------------------------------------------------------------------
// Fast WMMA GEMM: C[M,N] = X[M,K]*W[N,K]^T + bias (opt relu)
// Requires M%128==0, N%64==0, K%32==0. 8 waves, wave tile 16Mx64N.
// ---------------------------------------------------------------------------
__global__ __launch_bounds__(256)
void wmma_gemm_fast(const float* __restrict__ X, const float* __restrict__ W,
                    const float* __restrict__ bias, float* __restrict__ C,
                    int M, int N, int K, int relu) {
  int wave = threadIdx.x >> 5;
  int lane = threadIdx.x & 31;
  int m0 = blockIdx.y * 128 + wave * 16;
  int n0 = blockIdx.x * 64;
  int arow = m0 + (lane & 15);
  int kA = (lane >> 4) << 3;
  int kB = (lane >> 4) << 3;
  int bcol = n0 + (lane & 15);

  v8f acc[4] = {};
  for (int k0 = 0; k0 < K; k0 += 32) {
    v16bf a  = load_fragA_fast(X, K, arow, k0 + kA);
    v16bf b0 = load_fragB_fast(W, K, bcol,      k0 + kB);
    v16bf b1 = load_fragB_fast(W, K, bcol + 16, k0 + kB);
    v16bf b2 = load_fragB_fast(W, K, bcol + 32, k0 + kB);
    v16bf b3 = load_fragB_fast(W, K, bcol + 48, k0 + kB);
    acc[0] = __builtin_amdgcn_wmma_f32_16x16x32_bf16(false, a, false, b0, (short)0, acc[0], false, false);
    acc[1] = __builtin_amdgcn_wmma_f32_16x16x32_bf16(false, a, false, b1, (short)0, acc[1], false, false);
    acc[2] = __builtin_amdgcn_wmma_f32_16x16x32_bf16(false, a, false, b2, (short)0, acc[2], false, false);
    acc[3] = __builtin_amdgcn_wmma_f32_16x16x32_bf16(false, a, false, b3, (short)0, acc[3], false, false);
  }

  int rbase = m0 + ((lane >> 4) << 3);
  #pragma unroll
  for (int t = 0; t < 4; ++t) {
    int col = n0 + (lane & 15) + 16 * t;
    float bv = bias[col];
    #pragma unroll
    for (int r = 0; r < 8; ++r) {
      float v = acc[t][r] + bv;
      if (relu) v = fmaxf(v, 0.f);
      C[(size_t)(rbase + r) * N + col] = v;
    }
  }
}

// ---------------------------------------------------------------------------
// Edge WMMA GEMM for small/ragged shapes. Requires K%8==0.
// ---------------------------------------------------------------------------
__global__ __launch_bounds__(256)
void wmma_gemm_edge(const float* __restrict__ X, const float* __restrict__ W,
                    const float* __restrict__ bias, float* __restrict__ C,
                    int M, int N, int K, int relu) {
  int wave = threadIdx.x >> 5;
  int lane = threadIdx.x & 31;
  int m0 = blockIdx.y * 128 + wave * 16;
  int n0 = blockIdx.x * 64;
  int arow = m0 + (lane & 15);
  int kA = (lane >> 4) << 3;
  int kB = (lane >> 4) << 3;
  int bcol = n0 + (lane & 15);

  v8f acc[4] = {};
  for (int k0 = 0; k0 < K; k0 += 32) {
    v16bf a  = load_fragA_edge(X, M, K, arow, k0 + kA);
    v16bf b0 = load_fragB_edge(W, N, K, bcol,      k0 + kB);
    v16bf b1 = load_fragB_edge(W, N, K, bcol + 16, k0 + kB);
    v16bf b2 = load_fragB_edge(W, N, K, bcol + 32, k0 + kB);
    v16bf b3 = load_fragB_edge(W, N, K, bcol + 48, k0 + kB);
    acc[0] = __builtin_amdgcn_wmma_f32_16x16x32_bf16(false, a, false, b0, (short)0, acc[0], false, false);
    acc[1] = __builtin_amdgcn_wmma_f32_16x16x32_bf16(false, a, false, b1, (short)0, acc[1], false, false);
    acc[2] = __builtin_amdgcn_wmma_f32_16x16x32_bf16(false, a, false, b2, (short)0, acc[2], false, false);
    acc[3] = __builtin_amdgcn_wmma_f32_16x16x32_bf16(false, a, false, b3, (short)0, acc[3], false, false);
  }

  int rbase = m0 + ((lane >> 4) << 3);
  #pragma unroll
  for (int t = 0; t < 4; ++t) {
    int col = n0 + (lane & 15) + 16 * t;
    if (col < N) {
      float bv = bias ? bias[col] : 0.f;
      #pragma unroll
      for (int r = 0; r < 8; ++r) {
        int row = rbase + r;
        if (row < M) {
          float v = acc[t][r] + bv;
          if (relu) v = fmaxf(v, 0.f);
          C[(size_t)row * N + col] = v;
        }
      }
    }
  }
}

// ---------------------------------------------------------------------------
// Embedding: x[m,d] = emb[src[m], d] * sqrt(192)
// ---------------------------------------------------------------------------
__global__ void embed_kernel(const int* __restrict__ src,
                             const float* __restrict__ emb,
                             float* __restrict__ x) {
  size_t i = (size_t)blockIdx.x * blockDim.x + threadIdx.x;
  if (i >= (size_t)ML * DD) return;
  int d = (int)(i % DD);
  size_t m = i / DD;
  x[i] = emb[src[m] * DD + d] * 13.856406460551018f;
}

// ---------------------------------------------------------------------------
// Depthwise causal conv k=3, pad(2,0).  in: (B,L,576) -> out: (B,576,L)
// ---------------------------------------------------------------------------
__global__ void dwconv_kernel(const float* __restrict__ uc,
                              const float* __restrict__ w,
                              const float* __restrict__ bias,
                              float* __restrict__ out) {
  size_t i = (size_t)blockIdx.x * blockDim.x + threadIdx.x;
  if (i >= (size_t)BB * CIN * LL) return;
  int l = (int)(i % LL);
  size_t bc = i / LL;
  int c = (int)(bc % CIN);
  int b = (int)(bc / CIN);
  const float* in = uc + (size_t)b * LL * CIN + c;
  float acc = bias[c];
  #pragma unroll
  for (int t = 0; t < 3; ++t) {
    int ll = l + t - 2;
    if (ll >= 0) acc += w[c * 3 + t] * in[(size_t)ll * CIN];
  }
  out[i] = acc;
}

// ---------------------------------------------------------------------------
// Hyena implicit filter MLP: z(3) ->64 ->64 ->64 ->64, sin activations,
// exponential decay window.  block = (64 units, 4 positions)
// ---------------------------------------------------------------------------
__global__ __launch_bounds__(256)
void filter_kernel(const float* __restrict__ fw1, const float* __restrict__ fb1,
                   const float* __restrict__ fw2, const float* __restrict__ fb2,
                   const float* __restrict__ fw3, const float* __restrict__ fb3,
                   const float* __restrict__ fwo, const float* __restrict__ freq,
                   float* __restrict__ kfilt) {
  __shared__ float hs[4][64];
  __shared__ float hs2[4][64];
  int j = threadIdx.x;           // unit
  int ly = threadIdx.y;          // position slot
  int l = blockIdx.x * 4 + ly;   // 0..8191
  float t = (float)l / 34999.0f;                     // linspace(0,1,35000)[l]
  float wph = 6.283185307179586f * (float)l / 35000.0f;
  float phase = 1e-4f * wph;
  float z0 = t, z1 = __cosf(phase), z2 = -__sinf(phase);

  float a = fw1[j * 3] * z0 + fw1[j * 3 + 1] * z1 + fw1[j * 3 + 2] * z2 + fb1[j];
  hs[ly][j] = __sinf(freq[j] * a);
  __syncthreads();
  float acc = fb2[j];
  for (int k = 0; k < 64; ++k) acc += fw2[j * 64 + k] * hs[ly][k];
  hs2[ly][j] = __sinf(freq[j] * acc);
  __syncthreads();
  acc = fb3[j];
  for (int k = 0; k < 64; ++k) acc += fw3[j * 64 + k] * hs2[ly][k];
  __syncthreads();               // everyone done reading hs before overwrite
  hs[ly][j] = __sinf(freq[j] * acc);
  __syncthreads();
  acc = 0.f;
  for (int k = 0; k < 64; ++k) acc += fwo[j * 64 + k] * hs[ly][k];
  // deltas = linspace(ln(.01)/1.5, ln(.01)/0.3, 64); window = exp(-t*|delta|)
  float delta = -3.0701134573253944f +
                (-15.350567286626972f + 3.0701134573253944f) * (float)j / 63.0f;
  kfilt[(size_t)j * LL + l] = acc * __expf(-t * fabsf(delta));
}

// ---------------------------------------------------------------------------
// In-LDS radix-2 complex FFT, N=16384 (128KB LDS: needs CDNA5 320KB/WGP)
// ---------------------------------------------------------------------------
__device__ __forceinline__ void fft16384(float* re, float* im, int tid, int inv) {
  for (int i = tid; i < NSEQ; i += 1024) {
    int j = (int)(__brev((unsigned)i) >> 18);   // 14-bit reversal
    if (i < j) {
      float tr = re[i]; re[i] = re[j]; re[j] = tr;
      float ti = im[i]; im[i] = im[j]; im[j] = ti;
    }
  }
  __syncthreads();
  for (int s = 1; s <= 14; ++s) {
    int half = 1 << (s - 1);
    float ang0 = (inv ? 6.283185307179586f : -6.283185307179586f) / (float)(1 << s);
    for (int k = tid; k < NSEQ / 2; k += 1024) {
      int off = k & (half - 1);
      int base = ((k >> (s - 1)) << s) + off;
      float ang = ang0 * (float)off;
      float c = __cosf(ang), sn = __sinf(ang);
      float ur = re[base], ui = im[base];
      float vr = re[base + half], vi = im[base + half];
      float tr = vr * c - vi * sn;
      float ti = vr * sn + vi * c;
      re[base] = ur + tr;  im[base] = ui + ti;
      re[base + half] = ur - tr;  im[base + half] = ui - ti;
    }
    __syncthreads();
  }
}

// Spectrum of the implicit filter (per hd channel)
__global__ __launch_bounds__(1024)
void kfft_kernel(const float* __restrict__ kfilt, float2* __restrict__ Kf) {
  __shared__ float re[NSEQ];
  __shared__ float im[NSEQ];
  int hd = blockIdx.x;
  int tid = threadIdx.x;
  const float* src = kfilt + (size_t)hd * LL;
  for (int i = tid; i < NSEQ; i += 1024) {
    re[i] = (i < LL) ? src[i] : 0.f;
    im[i] = 0.f;
  }
  __syncthreads();
  fft16384(re, im, tid, 0);
  float2* dst = Kf + (size_t)hd * NSEQ;
  for (int i = tid; i < NSEQ; i += 1024) dst[i] = make_float2(re[i], im[i]);
}

// Fused: vx = v*x1 ; y = irfft(rfft(vx)*Kf)[:L] + vx*bias ; y *= x0 ;
// scatter into (B,L,D) layout for the output projection GEMM.
// Input staging uses gfx1250 async global->LDS DMA (ASYNCcnt-tracked).
__global__ __launch_bounds__(1024)
void fftconv_kernel(const float* __restrict__ convb, const float2* __restrict__ Kf,
                    const float* __restrict__ fbias, float* __restrict__ ybuf) {
  __shared__ float re[NSEQ];
  __shared__ float im[NSEQ];
  int s = blockIdx.x;                 // 0..3071
  int b = s / (HH * HD);
  int c = s % (HH * HD);
  int h = c >> 6;
  int hd = c & 63;
  const float* v  = convb + ((size_t)b * CIN + (size_t)(h * 3 + 2) * HD + hd) * LL;
  const float* x1 = convb + ((size_t)b * CIN + (size_t)(h * 3 + 1) * HD + hd) * LL;
  const float* x0 = convb + ((size_t)b * CIN + (size_t)(h * 3 + 0) * HD + hd) * LL;
  int tid = threadIdx.x;

  // async DMA: v -> im[0:LL], x1 -> im[LL:2*LL] (16B per lane, coalesced)
  for (int i = tid * 4; i < LL; i += 1024 * 4) {
    async_g2l_b128((unsigned)(size_t)&im[i],      v + i);
    async_g2l_b128((unsigned)(size_t)&im[LL + i], x1 + i);
  }
  WAIT_ASYNC0();
  __syncthreads();
  for (int i = tid; i < NSEQ; i += 1024)
    re[i] = (i < LL) ? im[i] * im[LL + i] : 0.f;
  __syncthreads();
  for (int i = tid; i < NSEQ; i += 1024) im[i] = 0.f;
  __syncthreads();

  fft16384(re, im, tid, 0);
  const float2* kf = Kf + (size_t)hd * NSEQ;
  for (int i = tid; i < NSEQ; i += 1024) {
    float2 w = kf[i];
    float ar = re[i], ai = im[i];
    re[i] = ar * w.x - ai * w.y;
    im[i] = ar * w.y + ai * w.x;
  }
  __syncthreads();
  fft16384(re, im, tid, 1);
  float bias = fbias[hd];
  for (int l = tid; l < LL; l += 1024) {
    float vx = v[l] * x1[l];
    float y = re[l] * (1.0f / (float)NSEQ) + vx * bias;
    y *= x0[l];
    ybuf[((size_t)b * LL + l) * DD + h * HD + hd] = y;
  }
}

// ---------------------------------------------------------------------------
// Residual-add + LayerNorm over D=192 (one wave per row)
// ---------------------------------------------------------------------------
__global__ __launch_bounds__(256)
void add_ln_kernel(const float* __restrict__ A, const float* __restrict__ B,
                   const float* __restrict__ g, const float* __restrict__ be,
                   float* __restrict__ out, int M) {
  int wid = (int)(((size_t)blockIdx.x * blockDim.x + threadIdx.x) >> 5);
  int lane = threadIdx.x & 31;
  if (wid >= M) return;
  const float* pa = A + (size_t)wid * DD;
  const float* pb = B + (size_t)wid * DD;
  float v[6];
  float s = 0.f, s2 = 0.f;
  #pragma unroll
  for (int i = 0; i < 6; ++i) {
    int d = lane + 32 * i;
    float x = pa[d] + pb[d];
    v[i] = x; s += x; s2 += x * x;
  }
  #pragma unroll
  for (int o = 16; o; o >>= 1) {
    s  += __shfl_xor(s,  o, 32);
    s2 += __shfl_xor(s2, o, 32);
  }
  float m = s * (1.f / DD);
  float var = s2 * (1.f / DD) - m * m;
  float inv = rsqrtf(var + 1e-5f);
  float* po = out + (size_t)wid * DD;
  #pragma unroll
  for (int i = 0; i < 6; ++i) {
    int d = lane + 32 * i;
    po[d] = (v[i] - m) * inv * g[d] + be[d];
  }
}

// reduced[m] = relu(sum_d out[m,d]*lin_w[d] + lin_b)  (one wave per row)
__global__ __launch_bounds__(256)
void rowdot_kernel(const float* __restrict__ X, const float* __restrict__ w,
                   const float* __restrict__ b, float* __restrict__ out, int M) {
  int wid = (int)(((size_t)blockIdx.x * blockDim.x + threadIdx.x) >> 5);
  int lane = threadIdx.x & 31;
  if (wid >= M) return;
  const float* row = X + (size_t)wid * DD;
  float s = 0.f;
  #pragma unroll
  for (int i = 0; i < 6; ++i) s += row[lane + 32 * i] * w[lane + 32 * i];
  #pragma unroll
  for (int o = 16; o; o >>= 1) s += __shfl_xor(s, o, 32);
  if (lane == 0) out[wid] = fmaxf(s + b[0], 0.f);
}

// BatchNorm over batch of 16 + relu (one thread per feature)
__global__ void bn_relu_kernel(const float* __restrict__ X, const float* __restrict__ g,
                               const float* __restrict__ b, float* __restrict__ out, int F) {
  int f = blockIdx.x * blockDim.x + threadIdx.x;
  if (f >= F) return;
  float s = 0.f, s2 = 0.f;
  for (int r = 0; r < BB; ++r) {
    float v = X[(size_t)r * F + f];
    s += v; s2 += v * v;
  }
  float m = s * (1.f / BB);
  float var = s2 * (1.f / BB) - m * m;
  float inv = rsqrtf(var + 1e-5f);
  for (int r = 0; r < BB; ++r)
    out[(size_t)r * F + f] = fmaxf((X[(size_t)r * F + f] - m) * inv * g[f] + b[f], 0.f);
}

// logits[r] = h2n[r,:] . fc3_w + fc3_b
__global__ void fc3_kernel(const float* __restrict__ X, const float* __restrict__ w,
                           const float* __restrict__ b, float* __restrict__ out) {
  int r = threadIdx.x;
  if (r < BB) {
    float s = b[0];
    for (int k = 0; k < 500; ++k) s += X[(size_t)r * 500 + k] * w[k];
    out[r] = s;
  }
}

// ---------------------------------------------------------------------------
// Host launch
// ---------------------------------------------------------------------------
// per-layer param indices (insertion order of the layer dict)
enum { IN_W=0, IN_B, SF_W, SF_B, FW1, FB1, FW2, FB2, FW3, FB3, FWO, FREQ,
       FBIAS, OUT_W, OUT_B, FF1_W, FF1_B, FF2_W, FF2_B, N1_G, N1_B, N2_G, N2_B,
       LAYER_NP };

extern "C" void kernel_launch(void* const* d_in, const int* in_sizes, int n_in,
                              void* d_out, int out_size, void* d_ws, size_t ws_size,
                              hipStream_t stream) {
  (void)in_sizes; (void)n_in; (void)out_size; (void)ws_size;
  const int* src = (const int*)d_in[0];
  const float* Pp[2][LAYER_NP];
  for (int l = 0; l < 2; ++l)
    for (int j = 0; j < LAYER_NP; ++j)
      Pp[l][j] = (const float*)d_in[1 + l * LAYER_NP + j];
  int base = 1 + 2 * LAYER_NP;
  const float* emb   = (const float*)d_in[base + 0];
  const float* lin_w = (const float*)d_in[base + 1];
  const float* lin_b = (const float*)d_in[base + 2];
  const float* fc1_w = (const float*)d_in[base + 3];
  const float* fc1_b = (const float*)d_in[base + 4];
  const float* bn1_g = (const float*)d_in[base + 5];
  const float* bn1_b = (const float*)d_in[base + 6];
  const float* fc2_w = (const float*)d_in[base + 7];
  const float* fc2_b = (const float*)d_in[base + 8];
  const float* bn2_g = (const float*)d_in[base + 9];
  const float* bn2_b = (const float*)d_in[base + 10];
  const float* fc3_w = (const float*)d_in[base + 11];
  const float* fc3_b = (const float*)d_in[base + 12];

  // output regions: [logits(16) | reduced(131072) | output(25165824)]
  float* out_logits  = (float*)d_out;
  float* out_reduced = out_logits + 16;
  float* out_x       = out_reduced + (size_t)ML;

  // workspace (~1.02 GB of f32)
  float* ws = (float*)d_ws;
  size_t o = 0;
  float* xbuf  = ws + o; o += (size_t)ML * DD;        // 25,165,824
  float* hbuf  = ws + o; o += (size_t)ML * DD;        // 25,165,824
  float* big1  = ws + o; o += (size_t)ML * 768;       // 100,663,296 (uc / ff1)
  float* convb = ws + o; o += (size_t)ML * CIN;       // 75,497,472
  float* ybuf  = ws + o; o += (size_t)ML * DD;        // 25,165,824
  float* kfilt = ws + o; o += (size_t)HD * LL;        // 524,288
  float* Kf    = ws + o; o += (size_t)HD * NSEQ * 2;  // 2,097,152
  float* h1    = ws + o; o += 16 * 1000;
  float* h1n   = ws + o; o += 16 * 1000;
  float* h2    = ws + o; o += 16 * 500;
  float* h2n   = ws + o; o += 16 * 500;

  // 1) embedding
  embed_kernel<<<(ML * DD + 255) / 256, 256, 0, stream>>>(src, emb, xbuf);

  // 2) encoder layers
  for (int l = 0; l < 2; ++l) {
    const float* const* P = Pp[l];
    // in-proj: (ML,192)x(576,192)^T -> big1   [aligned: fast path]
    wmma_gemm_fast<<<dim3(CIN / 64, ML / 128), 256, 0, stream>>>(
        xbuf, P[IN_W], P[IN_B], big1, ML, CIN, DD, 0);
    // depthwise causal conv -> convb (B,576,L)
    dwconv_kernel<<<((size_t)BB * CIN * LL + 255) / 256, 256, 0, stream>>>(
        big1, P[SF_W], P[SF_B], convb);
    // implicit filter + its spectrum
    filter_kernel<<<LL / 4, dim3(64, 4), 0, stream>>>(
        P[FW1], P[FB1], P[FW2], P[FB2], P[FW3], P[FB3], P[FWO], P[FREQ], kfilt);
    kfft_kernel<<<HD, 1024, 0, stream>>>(kfilt, (float2*)Kf);
    // fused gated FFT convolution -> ybuf (B,L,D)
    fftconv_kernel<<<BB * HH * HD, 1024, 0, stream>>>(
        convb, (const float2*)Kf, P[FBIAS], ybuf);
    // out-proj -> convb (reused as temp), then h = LN(x + proj)
    wmma_gemm_fast<<<dim3(DD / 64, ML / 128), 256, 0, stream>>>(
        ybuf, P[OUT_W], P[OUT_B], convb, ML, DD, DD, 0);
    add_ln_kernel<<<ML / 8, 256, 0, stream>>>(xbuf, convb, P[N1_G], P[N1_B], hbuf, ML);
    // FF: relu(h W1^T + b1) W2^T + b2
    wmma_gemm_fast<<<dim3(768 / 64, ML / 128), 256, 0, stream>>>(
        hbuf, P[FF1_W], P[FF1_B], big1, ML, 768, DD, 1);
    wmma_gemm_fast<<<dim3(DD / 64, ML / 128), 256, 0, stream>>>(
        big1, P[FF2_W], P[FF2_B], ybuf, ML, DD, 768, 0);
    float* xo = (l == 1) ? out_x : xbuf;   // final layer writes d_out directly
    add_ln_kernel<<<ML / 8, 256, 0, stream>>>(hbuf, ybuf, P[N2_G], P[N2_B], xo, ML);
  }

  // 3) head (ragged shapes -> edge kernel; K=8192 and K=1000 are both %8==0)
  rowdot_kernel<<<(ML * 32 + 255) / 256, 256, 0, stream>>>(out_x, lin_w, lin_b,
                                                           out_reduced, ML);
  wmma_gemm_edge<<<dim3((1000 + 63) / 64, 1), 256, 0, stream>>>(
      out_reduced, fc1_w, fc1_b, h1, BB, 1000, LL, 0);
  bn_relu_kernel<<<(1000 + 255) / 256, 256, 0, stream>>>(h1, bn1_g, bn1_b, h1n, 1000);
  wmma_gemm_edge<<<dim3((500 + 63) / 64, 1), 256, 0, stream>>>(
      h1n, fc2_w, fc2_b, h2, BB, 500, 1000, 0);
  bn_relu_kernel<<<(500 + 255) / 256, 256, 0, stream>>>(h2, bn2_g, bn2_b, h2n, 500);
  fc3_kernel<<<1, 32, 0, stream>>>(h2n, fc3_w, fc3_b, out_logits);
}
